// AttentionSecondOrderRNN_2113123910267
// MI455X (gfx1250) — matrix operand order repaired and checked by reading, not compiled, not run on hardware
//
#include <hip/hip_runtime.h>
#include <hip/hip_bf16.h>

// Problem dims (match reference)
#define B_  64
#define T_  2048
#define E_  256
#define H_  256
#define K_  4
#define KH_ 1024   // K_*H_

typedef __bf16 v16bf __attribute__((ext_vector_type(16)));
typedef __bf16 v8bf  __attribute__((ext_vector_type(8)));
typedef float  v8f   __attribute__((ext_vector_type(8)));

// ---------------------------------------------------------------------------
// f32 -> bf16 conversion (weights)
// ---------------------------------------------------------------------------
__global__ void cvt_f32_bf16_kernel(const float* __restrict__ src,
                                    __bf16* __restrict__ dst, int n) {
    int i = blockIdx.x * blockDim.x + threadIdx.x;
    if (i < n) dst[i] = (__bf16)src[i];
}

// ---------------------------------------------------------------------------
// alpha[b,t,k] = softmax_k( (x[b,t,:] . V[k,:]) / temperature )
// one thread per (b,t); V (4 KB) stays hot in cache
// ---------------------------------------------------------------------------
__global__ void alpha_kernel(const float* __restrict__ x,
                             const float* __restrict__ V,
                             const float* __restrict__ temperature,
                             float* __restrict__ alpha) {
    int idx = blockIdx.x * blockDim.x + threadIdx.x;   // = b*T_ + t
    const float* xr = x + (size_t)idx * E_;
    float acc[K_] = {0.f, 0.f, 0.f, 0.f};
    for (int e = 0; e < E_; e += 4) {
        float4 xv = *(const float4*)(xr + e);
#pragma unroll
        for (int k = 0; k < K_; ++k) {
            float4 vv = *(const float4*)(V + k * E_ + e);
            acc[k] += xv.x * vv.x + xv.y * vv.y + xv.z * vv.z + xv.w * vv.w;
        }
    }
    float inv_t = 1.0f / temperature[0];
    float m = -1e30f;
#pragma unroll
    for (int k = 0; k < K_; ++k) { acc[k] *= inv_t; m = fmaxf(m, acc[k]); }
    float s = 0.f;
#pragma unroll
    for (int k = 0; k < K_; ++k) { acc[k] = __expf(acc[k] - m); s += acc[k]; }
    float r = 1.0f / s;
#pragma unroll
    for (int k = 0; k < K_; ++k) alpha[(size_t)idx * K_ + k] = acc[k] * r;
}

// ---------------------------------------------------------------------------
// Phase 1: xw[t, b, kh] = sum_e x[b,t,e]*Wih[kh,e] + b_ih[kh] + b_hh[kh]  (bf16)
// Tile: rows = 16 batches (fixed t), cols = 64 kh per wave (4 WMMA tiles,
// A fragments reused across the 4 tiles). 8 waves / block.
// ---------------------------------------------------------------------------
__global__ __launch_bounds__(256)
void xw_gemm_kernel(const float* __restrict__ x,
                    const __bf16* __restrict__ wih,
                    const float* __restrict__ b_ih,
                    const float* __restrict__ b_hh,
                    __bf16* __restrict__ xw) {
    const int lane = threadIdx.x & 31;
    const int wave = threadIdx.x >> 5;
    const int job  = blockIdx.x * 8 + wave;      // T_*(B_/16)*(KH_/64) = 131072 jobs
    const int khg   = job & 15;                  // 64-column group
    const int btile = (job >> 4) & 3;            // 16-batch group
    const int t     = job >> 6;
    const int hi  = (lane >> 4) & 1;             // half-wave select
    const int col = lane & 15;

    const float* arow = x + ((size_t)(btile * 16 + col) * T_ + t) * E_ + hi * 16;
    const __bf16* brow = wih + (size_t)(khg * 64 + col) * E_ + hi * 16;

    v8f acc[4] = {v8f{}, v8f{}, v8f{}, v8f{}};
#pragma unroll
    for (int es = 0; es < 8; ++es) {
        // A fragment: elements 0..15 = K 16*hi .. 16*hi+15 (contiguous, matches B)
        const float* p = arow + es * 32;
        float4 f0 = *(const float4*)(p);
        float4 f1 = *(const float4*)(p + 4);
        float4 f2 = *(const float4*)(p + 8);
        float4 f3 = *(const float4*)(p + 12);
        v16bf a;
        a[0]=(__bf16)f0.x; a[1]=(__bf16)f0.y; a[2]=(__bf16)f0.z; a[3]=(__bf16)f0.w;
        a[4]=(__bf16)f1.x; a[5]=(__bf16)f1.y; a[6]=(__bf16)f1.z; a[7]=(__bf16)f1.w;
        a[8]=(__bf16)f2.x; a[9]=(__bf16)f2.y; a[10]=(__bf16)f2.z; a[11]=(__bf16)f2.w;
        a[12]=(__bf16)f3.x; a[13]=(__bf16)f3.y; a[14]=(__bf16)f3.z; a[15]=(__bf16)f3.w;
#pragma unroll
        for (int nt = 0; nt < 4; ++nt) {
            v16bf b = *(const v16bf*)(brow + (size_t)nt * 16 * E_ + es * 32);
            acc[nt] = __builtin_amdgcn_wmma_f32_16x16x32_bf16(
                false, a, false, b, (short)0, acc[nt], false, false);
        }
    }
#pragma unroll
    for (int nt = 0; nt < 4; ++nt) {
        const int kh = khg * 64 + nt * 16 + col;
        const float bias = b_ih[kh] + b_hh[kh];
        __bf16* orow = xw + ((size_t)t * B_ + btile * 16) * KH_ + kh;
#pragma unroll
        for (int j = 0; j < 8; ++j) {
            const int m = j + hi * 8;            // C layout: VGPR j -> M=j(+8)
            orow[(size_t)m * KH_] = (__bf16)(acc[nt][j] + bias);
        }
    }
}

// ---------------------------------------------------------------------------
// Async global->LDS staging of one 32 KB xw slab (contiguous), 32 B/thread.
// VDST VGPR = LDS byte offset = low 32 bits of the generic LDS pointer
// (flat-LDS aperture truncation, ISA 10.2). INST_OFFSET adds to BOTH sides.
// Each wave issues exactly 2 async ops -> s_wait_asynccnt 2 fences the
// previous stage while the new one stays in flight.
// ---------------------------------------------------------------------------
__device__ __forceinline__ void stage_async_slab(const __bf16* __restrict__ gsrc,
                                                 __bf16* ldst, int tid) {
    const char* g = (const char*)gsrc + (size_t)tid * 32;
    unsigned int l = (unsigned int)(unsigned long long)((char*)ldst + (size_t)tid * 32);
    asm volatile(
        "global_load_async_to_lds_b128 %0, %1, off\n\t"
        "global_load_async_to_lds_b128 %0, %1, off offset:16"
        :: "v"(l), "v"(g) : "memory");
}

// ---------------------------------------------------------------------------
// Phase 2: persistent scan. 4 workgroups x 1024 threads (32 waves); each
// owns 16 batch rows. Whh (bf16) register-resident: 2 col-tiles x 8 k-steps
// x v16bf per wave = 512 KB/WGP, zero weight traffic over 2048 steps.
// xw slabs double-buffered in LDS via async copies overlapping the WMMAs.
// ---------------------------------------------------------------------------
__global__ __launch_bounds__(1024)
void scan_kernel(const __bf16* __restrict__ xw,
                 const float* __restrict__ alpha,
                 const __bf16* __restrict__ whh,
                 float* __restrict__ out,        // [B,T,H]
                 float* __restrict__ hlast) {    // [B,H]
    __shared__ __bf16 sh_xw[2][16 * KH_];        // 2 x 32 KB double buffer
    __shared__ __bf16 sh_h[16 * H_];             // 8 KB  (h, bf16, row-major)
    __shared__ float  sh_acc[16 * H_];           // 16 KB (h_new accumulator)
    __shared__ float  sh_alpha[16 * K_];

    const int tid  = threadIdx.x;
    const int lane = tid & 31;
    const int wave = tid >> 5;
    const int bg   = blockIdx.x;                 // batch group (16 rows)
    const int hi   = (lane >> 4) & 1;
    const int col  = lane & 15;

    // Register-resident B fragments of Whh (columns kg, reduction h).
    v16bf bf[2][8];
#pragma unroll
    for (int nt = 0; nt < 2; ++nt) {
        const int kh = wave * 32 + nt * 16 + col;
        const __bf16* base = whh + (size_t)kh * H_ + hi * 16;
#pragma unroll
        for (int es = 0; es < 8; ++es)
            bf[nt][es] = *(const v16bf*)(base + es * 32);
    }

    const __bf16* xwbase = xw + (size_t)bg * 16 * KH_;   // slab t at + t*B_*KH_
    stage_async_slab(xwbase, sh_xw[0], tid);             // prefetch t=0

    for (int i = tid; i < 16 * H_; i += 1024) sh_h[i] = (__bf16)0.0f;  // h0 = 0
    __syncthreads();

    for (int t = 0; t < T_; ++t) {
        // -- phase 1: zero accum, load alpha, rotate async staging ----------
        {
            float4 z; z.x = z.y = z.z = z.w = 0.f;
            *(float4*)&sh_acc[tid * 4] = z;
        }
        if (tid < 64) {
            const int bl = tid >> 2, k = tid & 3;
            sh_alpha[tid] = alpha[((size_t)(bg * 16 + bl) * T_ + t) * K_ + k];
        }
        if (t + 1 < T_) {
            stage_async_slab(xwbase + (size_t)(t + 1) * B_ * KH_,
                             sh_xw[(t + 1) & 1], tid);
            asm volatile("s_wait_asynccnt 0x2" ::: "memory");  // slab t landed
        } else {
            asm volatile("s_wait_asynccnt 0x0" ::: "memory");
        }
        __syncthreads();

        // -- phase 2: hW = h x Whh^T (A from LDS, B resident in VGPRs) ------
        v8f c0 = v8f{}, c1 = v8f{};
#pragma unroll
        for (int es = 0; es < 8; ++es) {
            const __bf16* r = &sh_h[(size_t)col * H_ + es * 32 + hi * 16];
            v8bf lo = *(const v8bf*)(r);         // ds_load_b128
            v8bf hp = *(const v8bf*)(r + 8);     // ds_load_b128
            v16bf a = __builtin_shufflevector(lo, hp,
                0,1,2,3,4,5,6,7,8,9,10,11,12,13,14,15);
            c0 = __builtin_amdgcn_wmma_f32_16x16x32_bf16(
                false, a, false, bf[0][es], (short)0, c0, false, false);
            c1 = __builtin_amdgcn_wmma_f32_16x16x32_bf16(
                false, a, false, bf[1][es], (short)0, c1, false, false);
        }

        // h_cells = tanh(hW + xw); h_new[m,g] += alpha[m,k]*h_cells
        const __bf16* sxw = sh_xw[t & 1];
#pragma unroll
        for (int nt = 0; nt < 2; ++nt) {
            const int kh = wave * 32 + nt * 16 + col;
            const int k  = kh >> 8;
            const int g  = kh & (H_ - 1);
            v8f cc = nt ? c1 : c0;
#pragma unroll
            for (int j = 0; j < 8; ++j) {
                const int m = j + hi * 8;
                const float xv = (float)sxw[m * KH_ + kh];   // ds_load_u16
                const float hc = tanhf(cc[j] + xv);
                atomicAdd(&sh_acc[m * H_ + g], sh_alpha[m * K_ + k] * hc);
            }
        }
        __syncthreads();

        // -- phase 3: write combined_outputs[:, t, :], re-stage h as bf16 ---
        {
            const int i = tid * 4;
            const int m = i >> 8, g = i & (H_ - 1);
            float4 hv = *(float4*)&sh_acc[i];
            *(float4*)&out[((size_t)(bg * 16 + m) * T_ + t) * H_ + g] = hv;
            sh_h[i]     = (__bf16)hv.x;
            sh_h[i + 1] = (__bf16)hv.y;
            sh_h[i + 2] = (__bf16)hv.z;
            sh_h[i + 3] = (__bf16)hv.w;
        }
        __syncthreads();
    }

    // h_last
    {
        const int i = tid * 4;
        const int m = i >> 8, g = i & (H_ - 1);
        *(float4*)&hlast[(size_t)(bg * 16 + m) * H_ + g] = *(float4*)&sh_acc[i];
    }
}

// ---------------------------------------------------------------------------
extern "C" void kernel_launch(void* const* d_in, const int* in_sizes, int n_in,
                              void* d_out, int out_size, void* d_ws, size_t ws_size,
                              hipStream_t stream) {
    const float* x    = (const float*)d_in[0];
    const float* temp = (const float*)d_in[1];
    const float* Wih  = (const float*)d_in[2];
    const float* Whh  = (const float*)d_in[3];
    const float* b_ih = (const float*)d_in[4];
    const float* b_hh = (const float*)d_in[5];
    const float* V    = (const float*)d_in[6];

    float* out   = (float*)d_out;
    float* hlast = out + (size_t)B_ * T_ * H_;

    // Workspace carve (256B-aligned):
    //   xw bf16 [T,B,KH] 268MB | alpha f32 [B,T,K] 2MB | wih_b 512KB | whh_b 512KB
    char* w = (char*)d_ws;
    size_t off = 0;
    __bf16* xw = (__bf16*)(w + off);      off += (size_t)B_ * T_ * KH_ * 2;
    float* alpha = (float*)(w + off);     off += (size_t)B_ * T_ * K_ * 4;
    __bf16* wih_b = (__bf16*)(w + off);   off += (size_t)K_ * H_ * E_ * 2;
    __bf16* whh_b = (__bf16*)(w + off);   off += (size_t)K_ * H_ * H_ * 2;
    if (ws_size < off) return;  // workspace too small; nothing safe to do

    const int nWih = K_ * H_ * E_;
    const int nWhh = K_ * H_ * H_;
    cvt_f32_bf16_kernel<<<(nWih + 255) / 256, 256, 0, stream>>>(Wih, wih_b, nWih);
    cvt_f32_bf16_kernel<<<(nWhh + 255) / 256, 256, 0, stream>>>(Whh, whh_b, nWhh);
    alpha_kernel<<<(B_ * T_) / 256, 256, 0, stream>>>(x, V, temp, alpha);
    xw_gemm_kernel<<<16384, 256, 0, stream>>>(x, wih_b, b_ih, b_hh, xw);
    scan_kernel<<<4, 1024, 0, stream>>>(xw, alpha, whh_b, out, hlast);
}
